// SGMoEBlock_8770323218990
// MI455X (gfx1250) — compile-verified, matrix operand order checked by reading, not run on hardware
//
#include <hip/hip_runtime.h>
#include <hip/hip_bf16.h>

// ---------------- problem constants ----------------
#define BS   8
#define C_   128     // channels == num experts
#define T_   512
#define ED_  64
#define E_   128
#define NT   (BS * C_)          // 1024 tokens
#define NELE ((size_t)NT * T_)  // 524288 elems per tensor side

typedef __attribute__((ext_vector_type(16))) _Float16     v16h;
typedef __attribute__((ext_vector_type(8)))  float        v8f;
typedef __attribute__((ext_vector_type(4)))  unsigned int v4u;
typedef __attribute__((ext_vector_type(8)))  int          v8i;
typedef __attribute__((ext_vector_type(4)))  int          v4i;

// ---------------- workspace layout (bytes) ----------------
static constexpr size_t CNT_OFF     = 0;                          // int[128], pad to 1KB
static constexpr size_t ENTRIES_OFF = 1024;                       // int[128*1024]
static constexpr size_t EPROB_OFF   = ENTRIES_OFF + (size_t)E_ * NT * 4;
static constexpr size_t PARTIAL_OFF = EPROB_OFF   + (size_t)E_ * NT * 4;
// partial: float[2 sides][2 kslots][NT][T] = 8 MB
static constexpr size_t XH_OFF      = PARTIAL_OFF + (size_t)2 * 2 * NT * T_ * 4;
// xh: f16[2*NT][T] = 2 MB

__device__ __forceinline__ unsigned rl_u32(unsigned x) {
    return (unsigned)__builtin_amdgcn_readfirstlane((int)x);
}

// =====================================================================
// Phase 0: convert x_l / x_r to f16 table xh[side*NT + tok][T] for TDM.
// =====================================================================
__global__ __launch_bounds__(256)
void convert_x_kernel(const float* __restrict__ xl, const float* __restrict__ xr,
                      _Float16* __restrict__ xh)
{
    size_t i = (size_t)blockIdx.x * 256 + threadIdx.x;   // grid covers NELE
    xh[i]        = (_Float16)xl[i];
    xh[NELE + i] = (_Float16)xr[i];
}

// =====================================================================
// Phase 1: cosine router. One block per token (128 threads).
// =====================================================================
__global__ __launch_bounds__(128)
void router_kernel(const float* __restrict__ xl, const float* __restrict__ xr,
                   const float* __restrict__ centers,
                   const float* __restrict__ projW, const float* __restrict__ projb,
                   int* __restrict__ cnt, int* __restrict__ entries,
                   float* __restrict__ eprob)
{
    const int t   = blockIdx.x;      // token id = b*C + c
    const int tid = threadIdx.x;

    __shared__ float xproj[ED_];
    __shared__ float sims[E_];
    __shared__ float invn;

    const float* xl_t = xl + (size_t)t * T_;
    const float* xr_t = xr + (size_t)t * T_;

    if (tid < ED_) {
        const float* w = projW + (size_t)tid * (2 * T_);
        float acc = projb[tid];
        for (int s = 0; s < T_; ++s) acc = fmaf(xl_t[s], w[s], acc);
        for (int s = 0; s < T_; ++s) acc = fmaf(xr_t[s], w[T_ + s], acc);
        xproj[tid] = acc;
    }
    __syncthreads();

    if (tid == 0) {
        float ss = 0.f;
        for (int i = 0; i < ED_; ++i) ss = fmaf(xproj[i], xproj[i], ss);
        invn = 1.0f / fmaxf(sqrtf(ss), 1e-12f);
    }
    __syncthreads();

    {
        const float* ce = centers + (size_t)tid * ED_;
        float cs = 0.f, cn = 0.f;
        for (int i = 0; i < ED_; ++i) {
            float cv = ce[i];
            cs = fmaf(cv, xproj[i], cs);
            cn = fmaf(cv, cv, cn);
        }
        sims[tid] = cs * invn / fmaxf(sqrtf(cn), 1e-12f);
    }
    __syncthreads();

    if (tid == 0) {
        int   i0 = 0;    float s0 = sims[0];
        for (int i = 1; i < E_; ++i) if (sims[i] > s0) { s0 = sims[i]; i0 = i; }
        int   i1 = (i0 == 0) ? 1 : 0; float s1 = sims[i1];
        for (int i = 0; i < E_; ++i) if (i != i0 && sims[i] > s1) { s1 = sims[i]; i1 = i; }
        float m  = fmaxf(s0, s1);
        float e0 = __expf(s0 - m), e1 = __expf(s1 - m);
        float r  = 1.0f / (e0 + e1);

        int slot0 = atomicAdd(&cnt[i0], 1);
        entries[i0 * NT + slot0] = (t << 1) | 0;
        eprob  [i0 * NT + slot0] = e0 * r;
        int slot1 = atomicAdd(&cnt[i1], 1);
        entries[i1 * NT + slot1] = (t << 1) | 1;
        eprob  [i1 * NT + slot1] = e1 * r;
    }
}

// =====================================================================
// Phase 2: grouped expert GEMM. One block per expert (256 thr = 8 waves).
// B tile (16 gathered x-columns, f16) is DMA'd into LDS by the Tensor
// Data Mover in gather mode; A tiles (W_e rows) stream f32->f16 in regs;
// v_wmma_f32_16x16x32_f16 accumulates in f32.
// =====================================================================
__global__ __launch_bounds__(256)
void expert_kernel(const _Float16* __restrict__ xh,
                   const float* __restrict__ expW, const float* __restrict__ expb,
                   const int* __restrict__ cnt, const int* __restrict__ entries,
                   const float* __restrict__ eprob, float* __restrict__ partial)
{
    const int e = blockIdx.x;
    const int n = cnt[e];            // tokens routed to this expert
    const int ncols = 2 * n;         // each token contributes l and r columns
    if (ncols == 0) return;          // uniform across block

    const float* We = expW + (size_t)e * T_ * T_;
    const float* be = expb + (size_t)e * T_;

    __shared__ __align__(16) _Float16 Bl[16 * T_];   // 16 columns x 512 K (16 KB)
    __shared__ int            col_tok[16];
    __shared__ int            col_ks[16];
    __shared__ int            col_side[16];
    __shared__ float          col_p[16];
    __shared__ unsigned short rowidx16[16];          // gather row indices

    const int tid   = threadIdx.x;
    const int lane  = tid & 31;
    const int wave  = tid >> 5;
    const int khalf = lane >> 4;     // K-half select per ISA 16-bit A/B layout
    const int lsub  = lane & 15;     // row (A) / column (B,C) index

    const unsigned lds_base = (unsigned)(size_t)(&Bl[0]);       // LDS byte offset
    const unsigned long long ga = (unsigned long long)(size_t)xh;

    for (int ct = 0; ct * 16 < ncols; ++ct) {
        // -------- column metadata + gather row indices --------
        if (tid < 16) {
            int cg = ct * 16 + tid;
            if (cg < ncols) {
                int ent = entries[e * NT + (cg >> 1)];
                int tok = ent >> 1;
                int sd  = cg & 1;
                col_tok[tid]  = tok;
                col_ks[tid]   = ent & 1;
                col_side[tid] = sd;
                col_p[tid]    = eprob[e * NT + (cg >> 1)];
                rowidx16[tid] = (unsigned short)(sd * NT + tok); // row in xh
            } else {
                col_tok[tid] = -1; col_ks[tid] = 0; col_side[tid] = 0; col_p[tid] = 0.f;
                rowidx16[tid] = 0;                                // harmless garbage col
            }
        }
        __syncthreads();

        // -------- TDM gather: 16 rows x 512 f16 from xh -> LDS --------
        if (wave == 0) {
            // D# group 0: count=1, gather_mode=1 (16-bit indices), lds, global addr, type=2
            v4u g0;
            g0[0] = rl_u32(1u | (1u << 31));
            g0[1] = rl_u32(lds_base);
            g0[2] = rl_u32((unsigned)ga);
            g0[3] = rl_u32((unsigned)((ga >> 32) & 0x01ffffffull) | (2u << 30));
            // D# group 1: data_size=2B, tensor_dim0=512, tensor_dim1=2048,
            //             tile_dim0=512, tile_dim1=16 (#indices), dim0_stride=512
            v8i g1;
            g1[0] = (int)rl_u32(1u << 16);          // data_size=1 (2 bytes), mask=0
            g1[1] = (int)rl_u32(512u << 16);        // tensor_dim0[15:0] @ bits63:48
            g1[2] = (int)rl_u32(2048u << 16);       // tensor_dim1[15:0] @ bits95:80
            g1[3] = (int)rl_u32(512u << 16);        // tile_dim0 @ bits127:112
            g1[4] = (int)rl_u32(16u);               // tile_dim1 = #valid indices
            g1[5] = (int)rl_u32(512u);              // tensor_dim0_stride
            g1[6] = 0;
            g1[7] = 0;
            // D# groups 2/3: sixteen 16-bit row indices
            v4i g2, g3;
            #pragma unroll
            for (int j = 0; j < 4; ++j) {
                g2[j] = (int)rl_u32((unsigned)rowidx16[2 * j] |
                                    ((unsigned)rowidx16[2 * j + 1] << 16));
                g3[j] = (int)rl_u32((unsigned)rowidx16[8 + 2 * j] |
                                    ((unsigned)rowidx16[8 + 2 * j + 1] << 16));
            }
#if defined(__clang_major__) && (__clang_major__ >= 23)
            v8i gz = {0, 0, 0, 0, 0, 0, 0, 0};
            __builtin_amdgcn_tensor_load_to_lds(g0, g1, g2, g3, gz, 0);
#else
            __builtin_amdgcn_tensor_load_to_lds(g0, g1, g2, g3, 0);
#endif
            __builtin_amdgcn_s_wait_tensorcnt(0);
        }
        __syncthreads();

        // -------- each wave owns 4 row-tiles (rows wave*64 .. +63) --------
        for (int rt = 0; rt < 4; ++rt) {
            const int rowbase = (wave * 4 + rt) * 16;
            const int arow    = rowbase + lsub;
            const float* wrow = We + (size_t)arow * T_;
            __builtin_prefetch(wrow, 0, 3);     // global_prefetch_b8

            v8f acc = {};
            for (int kc = 0; kc < T_ / 32; ++kc) {
                const int kbase = kc * 32;
                v16h a, b;
                #pragma unroll
                for (int v = 0; v < 8; ++v) {
                    // ISA 16-bit A 16x32 layout:
                    //  lanes0-15: V0..3 -> K0..7, V4..7 -> K16..23 ; lanes16-31: +8
                    int kk = ((v < 4) ? (v * 2) : (16 + (v - 4) * 2)) + khalf * 8;
                    float2 wv = *(const float2*)(wrow + kbase + kk);
                    a[2 * v]     = (_Float16)wv.x;
                    a[2 * v + 1] = (_Float16)wv.y;
                    const _Float16* bp = &Bl[lsub * T_ + kbase + kk];
                    b[2 * v]     = bp[0];
                    b[2 * v + 1] = bp[1];
                }
                acc = __builtin_amdgcn_wmma_f32_16x16x32_f16(
                        /*neg_a=*/false, a, /*neg_b=*/false, b,
                        /*c_mod=*/(short)0, acc, /*reuse_a=*/false, /*reuse_b=*/false);
            }

            // C layout: lane -> col = lsub, row = rowbase + v + 8*khalf
            int tok = col_tok[lsub];
            if (tok >= 0) {
                float p   = col_p[lsub];
                int   ks  = col_ks[lsub];
                int   sd  = col_side[lsub];
                float* dst = partial + (((size_t)sd * 2 + ks) * NT + tok) * T_;
                #pragma unroll
                for (int v = 0; v < 8; ++v) {
                    int r = rowbase + v + khalf * 8;
                    dst[r] = p * (acc[v] + be[r]);
                }
            }
        }
        __syncthreads();
    }
}

// =====================================================================
// Phase 3: combine k-slots, LayerNorm (biased var), residual, write out.
// =====================================================================
__global__ __launch_bounds__(256)
void combine_kernel(const float* __restrict__ xl, const float* __restrict__ xr,
                    const float* __restrict__ partial,
                    const float* __restrict__ lnw_l, const float* __restrict__ lnb_l,
                    const float* __restrict__ lnw_r, const float* __restrict__ lnb_r,
                    float* __restrict__ out)
{
    const int bid  = blockIdx.x;        // side*NT + token
    const int side = bid >> 10;
    const int tok  = bid & (NT - 1);
    const int tid  = threadIdx.x;

    const float* p0 = partial + (((size_t)side * 2 + 0) * NT + tok) * T_;
    const float* p1 = partial + (((size_t)side * 2 + 1) * NT + tok) * T_;
    const float* x  = (side ? xr : xl) + (size_t)tok * T_;
    const float* w  = side ? lnw_r : lnw_l;
    const float* bb = side ? lnb_r : lnb_l;

    __shared__ float red[256];
    __shared__ float stats[2];

    float v0 = p0[tid]       + p1[tid];
    float v1 = p0[tid + 256] + p1[tid + 256];

    red[tid] = v0 + v1;
    __syncthreads();
    for (int s = 128; s > 0; s >>= 1) {
        if (tid < s) red[tid] += red[tid + s];
        __syncthreads();
    }
    if (tid == 0) stats[0] = red[0] * (1.0f / T_);
    __syncthreads();
    const float mean = stats[0];

    float d0 = v0 - mean, d1 = v1 - mean;
    red[tid] = d0 * d0 + d1 * d1;
    __syncthreads();
    for (int s = 128; s > 0; s >>= 1) {
        if (tid < s) red[tid] += red[tid + s];
        __syncthreads();
    }
    if (tid == 0) stats[1] = red[0] * (1.0f / T_);
    __syncthreads();
    const float rstd = rsqrtf(stats[1] + 1e-5f);

    float* o = out + (size_t)side * NELE + (size_t)tok * T_;
    o[tid]       = d0 * rstd * w[tid]       + bb[tid]       + x[tid];
    o[tid + 256] = d1 * rstd * w[tid + 256] + bb[tid + 256] + x[tid + 256];
}

// =====================================================================
extern "C" void kernel_launch(void* const* d_in, const int* in_sizes, int n_in,
                              void* d_out, int out_size, void* d_ws, size_t ws_size,
                              hipStream_t stream)
{
    const float* xl      = (const float*)d_in[0];
    const float* xr      = (const float*)d_in[1];
    const float* centers = (const float*)d_in[2];
    const float* projW   = (const float*)d_in[3];
    const float* projb   = (const float*)d_in[4];
    const float* expW    = (const float*)d_in[5];
    const float* expb    = (const float*)d_in[6];
    const float* lnw_l   = (const float*)d_in[7];
    const float* lnb_l   = (const float*)d_in[8];
    const float* lnw_r   = (const float*)d_in[9];
    const float* lnb_r   = (const float*)d_in[10];

    char*     ws      = (char*)d_ws;
    int*      cnt     = (int*)      (ws + CNT_OFF);
    int*      entries = (int*)      (ws + ENTRIES_OFF);
    float*    eprob   = (float*)    (ws + EPROB_OFF);
    float*    partial = (float*)    (ws + PARTIAL_OFF);
    _Float16* xh      = (_Float16*) (ws + XH_OFF);

    hipMemsetAsync(cnt, 0, 1024, stream);

    convert_x_kernel<<<(int)(NELE / 256), 256, 0, stream>>>(xl, xr, xh);
    router_kernel  <<<NT,  128, 0, stream>>>(xl, xr, centers, projW, projb,
                                             cnt, entries, eprob);
    expert_kernel  <<<E_,  256, 0, stream>>>(xh, expW, expb,
                                             cnt, entries, eprob, partial);
    combine_kernel <<<2 * NT, 256, 0, stream>>>(xl, xr, partial,
                                                lnw_l, lnb_l, lnw_r, lnb_r,
                                                (float*)d_out);
}
